// Self_Attention_39513699123438
// MI455X (gfx1250) — matrix operand order repaired and checked by reading, not compiled, not run on hardware
//
#include <hip/hip_runtime.h>

// Self-attention fused pipeline for MI455X (gfx1250, wave32):
//   k1: fused Q/K/V projection (x tile staged once, multiplied vs 3 weight
//       tiles -> 3x less activation traffic), fp32 in -> bf16 out, V transposed
//   k2: causal flash-attention, K/V blocks staged in LDS per workgroup via
//       async-DMA (ASYNCcnt), bf16 WMMA matmuls, fp32 softmax.
// Bandwidth-bound problem (~120 GFLOP vs ~200MB @ 23.3 TB/s): bf16
// intermediates, no materialized LxL scores, single pass over K/V.

#ifndef __has_builtin
#define __has_builtin(x) 0
#endif

#if __has_builtin(__builtin_amdgcn_global_load_async_to_lds_b128) && \
    __has_builtin(__builtin_amdgcn_s_wait_asynccnt)
#define HAVE_ASYNC_LDS 1
#define AS1 __attribute__((address_space(1)))
#define AS3 __attribute__((address_space(3)))
#else
#define HAVE_ASYNC_LDS 0
#endif

constexpr int Bb  = 4;
constexpr int Ls  = 2048;
constexpr int Dm  = 1024;
constexpr int Hh  = 16;
constexpr int DHd = 64;
constexpr int BL  = Bb * Ls;           // 8192
constexpr size_t QKV_ELEMS = (size_t)Bb * Hh * Ls * DHd;  // 8,388,608 bf16 each

typedef __attribute__((ext_vector_type(16))) __bf16 v16bf;
typedef __attribute__((ext_vector_type(8)))  float  v8f;
typedef __attribute__((ext_vector_type(4)))  int    v4i;

union Frag16 { v16bf v; unsigned u[8]; };

__device__ __forceinline__ unsigned short f2bf(float f) {
  unsigned u = __float_as_uint(f);
  u += 0x7FFFu + ((u >> 16) & 1u);     // round-to-nearest-even
  return (unsigned short)(u >> 16);
}
__device__ __forceinline__ unsigned pack2(float lo, float hi) {
  return (unsigned)f2bf(lo) | ((unsigned)f2bf(hi) << 16);
}
// 16-bit A-matrix 16x32 layout (ISA 7.12.2): VGPR j, lane-half lh -> k index
__device__ __forceinline__ int a_k(int j, int lh) {
  return ((j >> 2) << 4) + (lh << 3) + ((j & 3) << 1);
}
// 16-bit B-matrix 32x16 layout: VGPR j, lane-half lh -> k index
__device__ __forceinline__ int b_k(int j, int lh) {
  return ((j >> 2) << 4) + (lh << 3) + ((j & 3) << 1);
}

__device__ __forceinline__ v8f wmma_bf16(const v16bf& a, const v16bf& b, const v8f& c) {
  return __builtin_amdgcn_wmma_f32_16x16x32_bf16(
      false, a, false, b, (short)0, c, false, false);
}

// Stage 16 bytes global -> LDS, async DMA if the toolchain exposes it.
__device__ __forceinline__ void stage16(const unsigned short* gp, unsigned short* lp) {
#if HAVE_ASYNC_LDS
  __builtin_amdgcn_global_load_async_to_lds_b128(
      (AS1 v4i*)(const void*)gp, (AS3 v4i*)(void*)lp, 0, 0);
#else
  *(uint4*)lp = *(const uint4*)gp;
#endif
}
__device__ __forceinline__ void stage_wait() {
#if HAVE_ASYNC_LDS
  __builtin_amdgcn_s_wait_asynccnt(0);
#endif
}

// ---------------------------------------------------------------------------
// Kernel 1: fused Q/K/V projection.  grid = (BL/128, Dm/64), block = 256.
// Each block: 128x64 output tile for ALL THREE projections; the x tile is
// staged into LDS once per k-step and reused against Wq/Wk/Wv tiles.
// Q/K stored [B,H,L,DH] bf16, V stored transposed [B,H,DH,L] bf16.
// ---------------------------------------------------------------------------
__global__ __launch_bounds__(256) void qkv_proj_fused(
    const float* __restrict__ x,
    const float* __restrict__ Wq, const float* __restrict__ bq,
    const float* __restrict__ Wk, const float* __restrict__ bk,
    const float* __restrict__ Wv, const float* __restrict__ bv,
    unsigned short* __restrict__ qo,
    unsigned short* __restrict__ ko,
    unsigned short* __restrict__ vo)
{
  const int m0 = blockIdx.x * 128;
  const int n0 = blockIdx.y * 64;

  __shared__ unsigned short ldsA[128 * 40];      // x tile, row stride 40 (pad)
  __shared__ unsigned short ldsB[3][64 * 40];    // W tiles transposed [n][k]

  const int tid  = threadIdx.x;
  const int wave = tid >> 5;
  const int t    = tid & 31;
  const int lh   = t >> 4;
  const int tn   = t & 15;
  const int wr   = wave >> 1;   // 0..3  (32-row subtile)
  const int wc   = wave & 1;    // 0..1  (32-col subtile)

  const float* Ws[3]     = {Wq, Wk, Wv};
  const float* biases[3] = {bq, bk, bv};

  const v8f vzero = {0.f, 0.f, 0.f, 0.f, 0.f, 0.f, 0.f, 0.f};
  v8f c[3][2][2];
#pragma unroll
  for (int p = 0; p < 3; ++p)
#pragma unroll
    for (int i = 0; i < 2; ++i)
#pragma unroll
      for (int jj = 0; jj < 2; ++jj) c[p][i][jj] = vzero;

  for (int k0 = 0; k0 < Dm; k0 += 32) {
    __syncthreads();
    // Stage A tile (128x32 f32 -> bf16 LDS), float4 coalesced
#pragma unroll
    for (int i = 0; i < 4; ++i) {
      int idx4 = i * 256 + tid;
      int row  = idx4 >> 3;
      int col  = (idx4 & 7) << 2;
      const float4 v4 = *(const float4*)(x + (size_t)(m0 + row) * Dm + k0 + col);
      unsigned* dst = (unsigned*)&ldsA[row * 40 + col];
      dst[0] = pack2(v4.x, v4.y);
      dst[1] = pack2(v4.z, v4.w);
    }
    // Stage the three 32x64 W tiles transposed into LDS [n][k]
#pragma unroll
    for (int p = 0; p < 3; ++p) {
#pragma unroll
      for (int i = 0; i < 2; ++i) {
        int idx4 = i * 256 + tid;
        int kr   = idx4 >> 4;
        int col  = (idx4 & 15) << 2;
        const float4 v4 = *(const float4*)(Ws[p] + (size_t)(k0 + kr) * Dm + n0 + col);
        ldsB[p][(col + 0) * 40 + kr] = f2bf(v4.x);
        ldsB[p][(col + 1) * 40 + kr] = f2bf(v4.y);
        ldsB[p][(col + 2) * 40 + kr] = f2bf(v4.z);
        ldsB[p][(col + 3) * 40 + kr] = f2bf(v4.w);
      }
    }
    __syncthreads();

    Frag16 a[2];
#pragma unroll
    for (int i = 0; i < 2; ++i) {
      int arow = wr * 32 + i * 16 + tn;
#pragma unroll
      for (int j = 0; j < 8; ++j)
        a[i].u[j] = *(const unsigned*)&ldsA[arow * 40 + a_k(j, lh)];
    }
#pragma unroll
    for (int p = 0; p < 3; ++p) {
      Frag16 bf[2];
#pragma unroll
      for (int jj = 0; jj < 2; ++jj) {
        int brow = wc * 32 + jj * 16 + tn;
#pragma unroll
        for (int j = 0; j < 8; ++j)
          bf[jj].u[j] = *(const unsigned*)&ldsB[p][brow * 40 + b_k(j, lh)];
      }
#pragma unroll
      for (int i = 0; i < 2; ++i)
#pragma unroll
        for (int jj = 0; jj < 2; ++jj)
          c[p][i][jj] = wmma_bf16(a[i].v, bf[jj].v, c[p][i][jj]);
    }
  }

  // Epilogue: bias add + bf16 store into head-major layouts.
#pragma unroll
  for (int p = 0; p < 3; ++p) {
    float bias_v[2];
#pragma unroll
    for (int jj = 0; jj < 2; ++jj)
      bias_v[jj] = biases[p][n0 + wc * 32 + jj * 16 + tn];

#pragma unroll
    for (int i = 0; i < 2; ++i) {
#pragma unroll
      for (int jj = 0; jj < 2; ++jj) {
        int gc = n0 + wc * 32 + jj * 16 + tn;
        int h  = gc >> 6;
        int dh = gc & 63;
#pragma unroll
        for (int vj = 0; vj < 8; ++vj) {
          float val = c[p][i][jj][vj] + bias_v[jj];
          int gr = m0 + wr * 32 + i * 16 + vj + 8 * lh;   // global row (b*L + l)
          int b  = gr >> 11;
          int l  = gr & (Ls - 1);
          unsigned short o16 = f2bf(val);
          if (p < 2) {
            unsigned short* out = (p == 0) ? qo : ko;
            out[(((size_t)(b * Hh + h)) * Ls + l) * DHd + dh] = o16;
          } else {
            vo[(((size_t)(b * Hh + h)) * DHd + dh) * Ls + l] = o16;
          }
        }
      }
    }
  }
}

// ---------------------------------------------------------------------------
// Kernel 2: causal flash-attention.  grid = (B*H, L/64), block = 128 (4 waves)
// Workgroup stages each 32-key K/V block into LDS once (async 128-bit DMA,
// shared by all 4 waves); each wave owns 16 query rows with online softmax;
// matmuls via v_wmma_f32_16x16x32_bf16; fp32 stats/accum.
// ---------------------------------------------------------------------------
__global__ __launch_bounds__(128) void flash_attn(
    const unsigned short* __restrict__ q,
    const unsigned short* __restrict__ k,
    const unsigned short* __restrict__ vT,
    float* __restrict__ out)
{
  const int bh = blockIdx.x;          // b*H + h
  const int qt = blockIdx.y;          // 64-row query tile
  const int tid  = threadIdx.x;
  const int wave = tid >> 5;
  const int t    = tid & 31;
  const int lh   = t >> 4;
  const int tn   = t & 15;

  const size_t base = (size_t)bh * Ls * DHd;
  const int q0 = qt * 64 + wave * 16;

  __shared__ unsigned short ldsK[32 * 72];     // K block [key][dh], pad 64->72
  __shared__ unsigned short ldsV[64 * 40];     // V block [dh][key], pad 32->40
  __shared__ unsigned short ldsP[4][16 * 40];  // per-wave P scratch (C->A relayout)

  // Q fragments (16 rows x 64 dh) loaded once, bf16 A-layout.
  Frag16 qa[2];
#pragma unroll
  for (int f = 0; f < 2; ++f)
#pragma unroll
    for (int j = 0; j < 8; ++j)
      qa[f].u[j] = *(const unsigned*)&q[base + (size_t)(q0 + tn) * DHd + f * 32 + a_k(j, lh)];

  const v8f vzero = {0.f, 0.f, 0.f, 0.f, 0.f, 0.f, 0.f, 0.f};
  v8f o[4];
  o[0] = vzero; o[1] = vzero; o[2] = vzero; o[3] = vzero;
  float mrow[8], lrow[8];
#pragma unroll
  for (int j = 0; j < 8; ++j) { mrow[j] = -3.0e38f; lrow[j] = 0.f; }

  // Uniform causal block count for the whole workgroup (rows up to qt*64+63):
  const int nkv = 2 * qt + 2;
  for (int kvb = 0; kvb < nkv; ++kvb) {
    const int kv0 = kvb << 5;

    __syncthreads();   // previous iteration's LDS reads complete
    // --- stage K block: 32 keys x 64 dh (16B per thread x2, coalesced) ---
#pragma unroll
    for (int i = 0; i < 2; ++i) {
      int idx = i * 128 + tid;
      int row = idx >> 3;            // key in tile
      int c8  = (idx & 7) << 3;      // dh offset (x8 shorts)
      stage16(&k[base + (size_t)(kv0 + row) * DHd + c8], &ldsK[row * 72 + c8]);
    }
    // --- stage V block: 64 dh x 32 keys (transposed layout) ---
#pragma unroll
    for (int i = 0; i < 2; ++i) {
      int idx = i * 128 + tid;
      int row = idx >> 2;            // dh in tile
      int c8  = (idx & 3) << 3;      // key offset (x8 shorts)
      stage16(&vT[base + (size_t)row * Ls + kv0 + c8], &ldsV[row * 40 + c8]);
    }
    // Prefetch next block into L2 while we compute (global_prefetch_b8).
    if (kvb + 1 < nkv) {
      int idx = tid;
      int row = idx >> 3, c8 = (idx & 7) << 3;
      __builtin_prefetch(&k[base + (size_t)(kv0 + 32 + row) * DHd + c8], 0, 1);
      int vrow = idx >> 2, vc8 = (idx & 3) << 3;
      __builtin_prefetch(&vT[base + (size_t)vrow * Ls + kv0 + 32 + vc8], 0, 1);
    }
    stage_wait();
    __syncthreads();

    // --- S = Q * K^T for 32 keys (two 16x16 C tiles) ---
    v8f s0 = vzero, s1 = vzero;
#pragma unroll
    for (int f = 0; f < 2; ++f) {
      Frag16 kf0, kf1;
#pragma unroll
      for (int j = 0; j < 8; ++j) {
        kf0.u[j] = *(const unsigned*)&ldsK[(tn)      * 72 + f * 32 + b_k(j, lh)];
        kf1.u[j] = *(const unsigned*)&ldsK[(16 + tn) * 72 + f * 32 + b_k(j, lh)];
      }
      s0 = wmma_bf16(qa[f].v, kf0.v, s0);
      s1 = wmma_bf16(qa[f].v, kf1.v, s1);
    }

    // --- scale, causal mask, online softmax (row reductions via shfl_xor) ---
#pragma unroll
    for (int j = 0; j < 8; ++j) {
      const int m    = j + 8 * lh;
      const int qrow = q0 + m;
      float v0 = s0[j] * 0.125f;                    // 1/sqrt(DH)
      float v1 = s1[j] * 0.125f;
      if (kv0 + tn      > qrow) v0 = -3.0e38f;
      if (kv0 + 16 + tn > qrow) v1 = -3.0e38f;

      float tm = fmaxf(v0, v1);
      tm = fmaxf(tm, __shfl_xor(tm, 1));
      tm = fmaxf(tm, __shfl_xor(tm, 2));
      tm = fmaxf(tm, __shfl_xor(tm, 4));
      tm = fmaxf(tm, __shfl_xor(tm, 8));

      const float mn = fmaxf(mrow[j], tm);
      const float al = __expf(mrow[j] - mn);
      const float p0 = __expf(v0 - mn);
      const float p1 = __expf(v1 - mn);
      float rs = p0 + p1;
      rs += __shfl_xor(rs, 1);
      rs += __shfl_xor(rs, 2);
      rs += __shfl_xor(rs, 4);
      rs += __shfl_xor(rs, 8);

      lrow[j] = lrow[j] * al + rs;
      mrow[j] = mn;
      o[0][j] *= al; o[1][j] *= al; o[2][j] *= al; o[3][j] *= al;

      ldsP[wave][m * 40 + tn]      = f2bf(p0);
      ldsP[wave][m * 40 + 16 + tn] = f2bf(p1);
    }

    // --- P (C layout) -> A-fragment via per-wave LDS roundtrip ---
    Frag16 pa;
#pragma unroll
    for (int j = 0; j < 8; ++j)
      pa.u[j] = *(const unsigned*)&ldsP[wave][tn * 40 + a_k(j, lh)];

    // --- O += P * V  (V tile [dh][key] -> contiguous key pairs) ---
#pragma unroll
    for (int d = 0; d < 4; ++d) {
      Frag16 vf;
#pragma unroll
      for (int j = 0; j < 8; ++j)
        vf.u[j] = *(const unsigned*)&ldsV[(d * 16 + tn) * 40 + b_k(j, lh)];
      o[d] = wmma_bf16(pa.v, vf.v, o[d]);
    }
  }

  // --- normalize and write fp32 output [B, L, H*DH] ---
  const int b = bh >> 4;
  const int h = bh & 15;
#pragma unroll
  for (int j = 0; j < 8; ++j) {
    const int m = j + 8 * lh;
    const float inv = 1.0f / lrow[j];
    const size_t orow = ((size_t)b * Ls + (q0 + m)) * Dm + h * 64;
#pragma unroll
    for (int d = 0; d < 4; ++d)
      out[orow + d * 16 + tn] = o[d][j] * inv;
  }
}

// ---------------------------------------------------------------------------
extern "C" void kernel_launch(void* const* d_in, const int* in_sizes, int n_in,
                              void* d_out, int out_size, void* d_ws, size_t ws_size,
                              hipStream_t stream) {
  (void)in_sizes; (void)n_in; (void)out_size; (void)ws_size;
  const float* x  = (const float*)d_in[0];
  const float* Wq = (const float*)d_in[1];
  const float* bq = (const float*)d_in[2];
  const float* Wk = (const float*)d_in[3];
  const float* bk = (const float*)d_in[4];
  const float* Wv = (const float*)d_in[5];
  const float* bv = (const float*)d_in[6];
  // d_in[7] = atten_mask: strictly-causal triu(k=1), computed analytically.

  unsigned short* q_ws = (unsigned short*)d_ws;
  unsigned short* k_ws = q_ws + QKV_ELEMS;
  unsigned short* v_ws = k_ws + QKV_ELEMS;   // 48 MB total bf16 scratch

  dim3 g1(BL / 128, Dm / 64), b1(256);
  qkv_proj_fused<<<g1, b1, 0, stream>>>(x, Wq, bq, Wk, bk, Wv, bv, q_ws, k_ws, v_ws);

  dim3 g2(Bb * Hh, Ls / 64), b2(128);
  flash_attn<<<g2, b2, 0, stream>>>(q_ws, k_ws, v_ws, (float*)d_out);
}
